// Fuse2_609885356650
// MI455X (gfx1250) — matrix operand order repaired and checked by reading, not compiled, not run on hardware
//
#include <hip/hip_runtime.h>
#include <hip/hip_bf16.h>

// Fuse2 forward for MI455X (gfx1250, wave32, WMMA).
// - All conv/GEMM math via v_wmma_f32_16x16x32_f16 (f16 in, f32 accum).
// - 1x1 convs (m1, fus, fe): double-buffered LDS B-tiles filled with
//   global_load_async_to_lds_b128 (ASYNCcnt) + s_wait_asynccnt.
// - 3x3 convs (m0, m2, m4) + cb-carrying 1x1s (m3, m5): direct predicated
//   global loads.
// - Each wave computes a 16x(16*NT) tile; B fragments are preloaded into
//   distinct registers, with a sched_barrier keeping the load group ahead of
//   the WMMA group (loads clause; single wait; WMMAs issue back-to-back).
// Workspace usage ~450 MB.

typedef _Float16 v16h __attribute__((ext_vector_type(16)));
typedef _Float16 v8h  __attribute__((ext_vector_type(8)));
typedef float    v8f  __attribute__((ext_vector_type(8)));

#define BN_EPS 1e-5f

__device__ __forceinline__ v16h load16(const _Float16* p0, const _Float16* p1) {
  v8h a = *(const v8h*)p0;
  v8h b = *(const v8h*)p1;
  v16h r;
#pragma unroll
  for (int i = 0; i < 8; ++i) { r[i] = a[i]; r[i + 8] = b[i]; }
  return r;
}

// Async global->LDS 16-byte copy (gfx1250, tracked by ASYNCcnt).
// dst_lds: raw LDS byte address (low 32 bits of the flat shared address).
__device__ __forceinline__ void async_cp16(unsigned dst_lds, const void* src) {
  asm volatile("global_load_async_to_lds_b128 %0, %1, off"
               :: "v"(dst_lds), "v"(src) : "memory");
}

// ---------------------------------------------------------------------------
// Direct implicit-GEMM conv (square kernel KHxKH, stride/pad), f16 in/WMMA.
//   in  : NHWC f16 [B][H][W][Cin];  wpk : f16 [Cout][KH*KH][Cin]
// Wave w -> output rows [16w,16w+16); block tiles 16*NT output pixels.
// ---------------------------------------------------------------------------
template <int NT>
__global__ __launch_bounds__(256) void conv_wmma_direct(
    const _Float16* __restrict__ in, const _Float16* __restrict__ wpk,
    const float* __restrict__ cb,
    const float* __restrict__ g, const float* __restrict__ bb,
    const float* __restrict__ mm, const float* __restrict__ vv,
    _Float16* __restrict__ out_h, float* __restrict__ out_f,
    float* __restrict__ sp,
    int H, int W, int Cin, int Cout, int KH, int stride, int pad,
    int out_cs, int act /*0=relu,1=silu*/)
{
  __shared__ float spl[128];
  const int OW = (W + 2 * pad - KH) / stride + 1;
  const int OH = (H + 2 * pad - KH) / stride + 1;
  const int NP = OH * OW;
  const int wave  = threadIdx.x >> 5;
  const int lane  = threadIdx.x & 31;
  const int hi    = (lane >> 4) & 1;
  const int mBase = wave << 4;
  const int b     = blockIdx.y;
  const int pixBase = blockIdx.x * (16 * NT);
  const int KK = KH * KH;

  if (sp) { if (threadIdx.x < Cout) spl[threadIdx.x] = 0.f; __syncthreads(); }

  v8f acc[NT];
#pragma unroll
  for (int nt = 0; nt < NT; ++nt) acc[nt] = (v8f){};

  int ohs[NT], ows[NT];
#pragma unroll
  for (int nt = 0; nt < NT; ++nt) {
    int p = pixBase + nt * 16 + (lane & 15);
    ohs[nt] = p / OW; ows[nt] = p % OW;
  }

  const _Float16* arow = wpk + (size_t)(mBase + (lane & 15)) * KK * Cin;
#pragma unroll 1
  for (int t = 0; t < KK; ++t) {
    const int kh = t / KH, kw = t % KH;
    const _Float16* atap = arow + (size_t)t * Cin;
    for (int k0 = 0; k0 < Cin; k0 += 32) {
      const _Float16* ap = atap + k0 + (hi ? 8 : 0);
      v16h af = load16(ap, ap + 16);
      // Preload all NT B fragments (distinct regs -> loads clause together).
      v16h bf[NT];
#pragma unroll
      for (int nt = 0; nt < NT; ++nt) {
        const int ih = ohs[nt] * stride - pad + kh;
        const int iw = ows[nt] * stride - pad + kw;
        const bool ok = (ih >= 0) & (ih < H) & (iw >= 0) & (iw < W);
        bf[nt] = (v16h){};
        if (ok) {
          const _Float16* bp = in + ((size_t)b * H * W + (size_t)ih * W + iw) * Cin
                                  + k0 + (hi ? 8 : 0);
          bf[nt] = load16(bp, bp + 16);
        }
      }
      // Keep the load group ahead of the WMMA group.
      __builtin_amdgcn_sched_barrier(0);
#pragma unroll
      for (int nt = 0; nt < NT; ++nt)
        acc[nt] = __builtin_amdgcn_wmma_f32_16x16x32_f16(false, af, false, bf[nt],
                                                         (short)0, acc[nt], false, false);
    }
  }

#pragma unroll
  for (int j = 0; j < 8; ++j) {
    const int co = mBase + j + (hi ? 8 : 0);
    const float s  = g[co] * rsqrtf(vv[co] + BN_EPS);
    const float sh = bb[co] - mm[co] * s + (cb ? cb[co] * s : 0.f);
#pragma unroll
    for (int nt = 0; nt < NT; ++nt) {
      const int pix = pixBase + nt * 16 + (lane & 15);
      float x = acc[nt][j] * s + sh;
      x = (act == 1) ? (x / (1.f + expf(-x))) : fmaxf(x, 0.f);
      if (out_h) out_h[((size_t)b * NP + pix) * out_cs + co] = (_Float16)x;
      if (out_f) out_f[((size_t)b * Cout + co) * NP + pix] = x;
      if (sp) atomicAdd(&spl[co], x);
    }
  }
  if (sp) {
    __syncthreads();
    if (threadIdx.x < Cout) atomicAdd(&sp[b * Cout + threadIdx.x], spl[threadIdx.x]);
  }
}

// ---------------------------------------------------------------------------
// 1x1-conv GEMM with async-LDS double-buffered B tiles. 256 threads (8 waves,
// Cout=128). Block covers M=128 x N=64 pixels; K step = 32.
//   in : NHWC f16 [B][NP][Cin]; wpk : f16 [Cout][Cin]
// LDS B tile: [64 pixels][40 halves] (80B row stride for bank spread).
// Fill = one global_load_async_to_lds_b128 per thread per K-step (NFW=1/wave).
// ---------------------------------------------------------------------------
__global__ __launch_bounds__(256) void gemm_wmma_async(
    const _Float16* __restrict__ in, const _Float16* __restrict__ wpk,
    const float* __restrict__ g, const float* __restrict__ bb,
    const float* __restrict__ mm, const float* __restrict__ vv,
    _Float16* __restrict__ out_h, float* __restrict__ out_f,
    float* __restrict__ sp,
    int NP, int Cin, int Cout, int out_cs, int act)
{
  __shared__ __align__(16) _Float16 Bt[2][64 * 40];
  __shared__ float spl[128];
  const int tid   = threadIdx.x;
  const int wave  = tid >> 5;
  const int lane  = tid & 31;
  const int hi    = (lane >> 4) & 1;
  const int mBase = wave << 4;
  const int b     = blockIdx.y;
  const int pixBase = blockIdx.x * 64;
  const int KSTEPS = Cin >> 5;

  if (sp) { if (tid < Cout) spl[tid] = 0.f; }

  const _Float16* gin = in + (size_t)((size_t)b * NP + pixBase) * Cin;
  const unsigned lds0 = (unsigned)(uintptr_t)(void*)&Bt[0][0];
  const unsigned lds1 = (unsigned)(uintptr_t)(void*)&Bt[1][0];

  // This thread's 16B chunk within the 64x64B tile: pixel p, half-offset kh.
  const int fp  = tid >> 2;
  const int fkh = (tid & 3) * 8;
  const unsigned fdst = (unsigned)(fp * 80 + fkh * 2);
  const _Float16* fsrc = gin + (size_t)fp * Cin + fkh;

  async_cp16(lds0 + fdst, fsrc);  // prologue fill, K-step 0

  v8f acc[4];
#pragma unroll
  for (int nt = 0; nt < 4; ++nt) acc[nt] = (v8f){};

  const _Float16* arow = wpk + (size_t)(mBase + (lane & 15)) * Cin;
#pragma unroll 1
  for (int ks = 0; ks < KSTEPS; ++ks) {
    const int cur = ks & 1;
    const bool more = (ks + 1) < KSTEPS;
    if (more) async_cp16((cur ? lds0 : lds1) + fdst, fsrc + ((ks + 1) << 5));
    // Own fills of current buffer complete (next-buffer fill may remain).
    if (more) asm volatile("s_wait_asynccnt 0x1" ::: "memory");
    else      asm volatile("s_wait_asynccnt 0x0" ::: "memory");
    __syncthreads();              // all waves' fills of current buffer done

    const int k0 = ks << 5;
    const _Float16* ap = arow + k0 + (hi ? 8 : 0);
    v16h af = load16(ap, ap + 16);
    v16h bf[4];
#pragma unroll
    for (int nt = 0; nt < 4; ++nt) {
      const _Float16* bp = &Bt[cur][(nt * 16 + (lane & 15)) * 40 + (hi ? 8 : 0)];
      bf[nt] = load16(bp, bp + 16);
    }
    // Keep the 8 ds_load_b128 ahead of the WMMA group (distinct registers).
    __builtin_amdgcn_sched_barrier(0);
#pragma unroll
    for (int nt = 0; nt < 4; ++nt)
      acc[nt] = __builtin_amdgcn_wmma_f32_16x16x32_f16(false, af, false, bf[nt],
                                                       (short)0, acc[nt], false, false);
    if (more) __syncthreads();    // done reading before this buffer is refilled
  }

#pragma unroll
  for (int j = 0; j < 8; ++j) {
    const int co = mBase + j + (hi ? 8 : 0);
    const float s  = g[co] * rsqrtf(vv[co] + BN_EPS);
    const float sh = bb[co] - mm[co] * s;
#pragma unroll
    for (int nt = 0; nt < 4; ++nt) {
      const int pix = pixBase + nt * 16 + (lane & 15);
      float x = acc[nt][j] * s + sh;
      x = (act == 1) ? (x / (1.f + expf(-x))) : fmaxf(x, 0.f);
      if (out_h) out_h[((size_t)b * NP + pix) * out_cs + co] = (_Float16)x;
      if (out_f) out_f[((size_t)b * Cout + co) * NP + pix] = x;
      if (sp) atomicAdd(&spl[co], x);
    }
  }
  if (sp) {
    __syncthreads();
    if (tid < Cout) atomicAdd(&sp[b * Cout + tid], spl[tid]);
  }
}

// ---------------------------------------------------------------------------
// Weight repack: OIHW f32 -> [Cout][KK][Cin] f16 (KK=1 handles 1x1 weights).
// ---------------------------------------------------------------------------
__global__ void repack_w(const float* __restrict__ w, _Float16* __restrict__ o,
                         int Cout, int Cin, int KK) {
  int i = blockIdx.x * blockDim.x + threadIdx.x;
  int tot = Cout * Cin * KK;
  if (i >= tot) return;
  int co = i / (KK * Cin);
  int r  = i % (KK * Cin);
  int t  = r / Cin;
  int ci = r % Cin;
  o[i] = (_Float16)w[((size_t)co * Cin + ci) * KK + t];
}

// Pack concat(t, rgb) NHWC f16; also prefill wrapcat ch128-255=rgb and
// fecat ch128-255=t.
__global__ void pack_input(const float* __restrict__ rgb, const float* __restrict__ t,
                           _Float16* __restrict__ in_nhwc,
                           _Float16* __restrict__ wrapcat,
                           _Float16* __restrict__ fecat,
                           int B, int C, int NP) {
  size_t i = (size_t)blockIdx.x * blockDim.x + threadIdx.x;
  size_t tot = (size_t)B * C * NP;
  if (i >= tot) return;
  int pix = (int)(i % NP);
  int c   = (int)((i / NP) % C);
  int b   = (int)(i / ((size_t)NP * C));
  float tv = t[i], rv = rgb[i];
  size_t base = ((size_t)b * NP + pix) * (size_t)(2 * C);
  in_nhwc[base + c]     = (_Float16)tv;
  in_nhwc[base + C + c] = (_Float16)rv;
  wrapcat[base + C + c] = (_Float16)rv;
  fecat[base + C + c]   = (_Float16)tv;
}

// STN head: pooled mean (from sp sums) -> theta (f16-rounded) per batch.
// theta[b] = {t00, t02, t11, t12}.
__global__ void stn_head(const float* __restrict__ sp5,
                         const float* __restrict__ xy_w, const float* __restrict__ xy_b,
                         const float* __restrict__ s1_w, const float* __restrict__ s1_b,
                         const float* __restrict__ s2_w, const float* __restrict__ s2_b,
                         float* __restrict__ theta, int B, int Cp, float inv_np) {
  int b = threadIdx.x;
  if (b >= B) return;
  float sxy0 = xy_b[0], sxy1 = xy_b[1], ss1 = s1_b[0], ss2 = s2_b[0];
  for (int c = 0; c < Cp; ++c) {
    float p = sp5[b * Cp + c] * inv_np;
    sxy0 += xy_w[c] * p;
    sxy1 += xy_w[Cp + c] * p;
    ss1  += s1_w[c] * p;
    ss2  += s2_w[c] * p;
  }
  theta[b * 4 + 0] = (float)(_Float16)(1.f + ss1);
  theta[b * 4 + 1] = (float)(_Float16)sxy0;
  theta[b * 4 + 2] = (float)(_Float16)(1.f + ss2);
  theta[b * 4 + 3] = (float)(_Float16)sxy1;
}

// Affine bilinear grid sample of src (f32 NCHW) -> wrapcat ch 0..C-1 (f16 NHWC).
__global__ void grid_sample_k(const float* __restrict__ src,
                              const float* __restrict__ theta,
                              _Float16* __restrict__ wrapcat,
                              int B, int C, int H, int W) {
  int i = blockIdx.x * blockDim.x + threadIdx.x;
  int NP = H * W;
  if (i >= B * NP) return;
  int b = i / NP, pix = i % NP;
  int h = pix / W, w = pix % W;
  float xs = -1.f + 2.f * w / (float)(W - 1);
  float ys = -1.f + 2.f * h / (float)(H - 1);
  float gx = theta[b * 4 + 0] * xs + theta[b * 4 + 1];
  float gy = theta[b * 4 + 2] * ys + theta[b * 4 + 3];
  float ix = (gx + 1.f) * 0.5f * (float)(W - 1);
  float iy = (gy + 1.f) * 0.5f * (float)(H - 1);
  float ix0 = floorf(ix), iy0 = floorf(iy);
  float ix1 = ix0 + 1.f, iy1 = iy0 + 1.f;
  float wx1 = ix - ix0, wx0 = 1.f - wx1;
  float wy1 = iy - iy0, wy0 = 1.f - wy1;
  float v00 = ((iy0 >= 0.f) & (iy0 <= (float)(H - 1)) & (ix0 >= 0.f) & (ix0 <= (float)(W - 1))) ? 1.f : 0.f;
  float v01 = ((iy0 >= 0.f) & (iy0 <= (float)(H - 1)) & (ix1 >= 0.f) & (ix1 <= (float)(W - 1))) ? 1.f : 0.f;
  float v10 = ((iy1 >= 0.f) & (iy1 <= (float)(H - 1)) & (ix0 >= 0.f) & (ix0 <= (float)(W - 1))) ? 1.f : 0.f;
  float v11 = ((iy1 >= 0.f) & (iy1 <= (float)(H - 1)) & (ix1 >= 0.f) & (ix1 <= (float)(W - 1))) ? 1.f : 0.f;
  int x0 = min(max((int)ix0, 0), W - 1), x1c = min(max((int)ix1, 0), W - 1);
  int y0 = min(max((int)iy0, 0), H - 1), y1c = min(max((int)iy1, 0), H - 1);
  float w00 = wy0 * wx0 * v00, w01 = wy0 * wx1 * v01;
  float w10 = wy1 * wx0 * v10, w11 = wy1 * wx1 * v11;
  const float* sb = src + (size_t)b * C * NP;
  size_t obase = ((size_t)b * NP + pix) * (size_t)(2 * C);
  for (int c = 0; c < C; ++c) {
    const float* p = sb + (size_t)c * NP;
    float v = p[y0 * W + x0] * w00 + p[y0 * W + x1c] * w01 +
              p[y1c * W + x0] * w10 + p[y1c * W + x1c] * w11;
    wrapcat[obase + c] = (_Float16)v;
  }
}

// Dynamic-weight / attention heads from spatial mean of `shared`.
__global__ __launch_bounds__(256) void head2(const float* __restrict__ spfe,
    const float* __restrict__ wg1, const float* __restrict__ wg2,
    const float* __restrict__ ag1, const float* __restrict__ ag2,
    float* __restrict__ cw, float* __restrict__ attn, float inv_np) {
  __shared__ float spm[128], t1[32], ta[32];
  int b = blockIdx.x, tid = threadIdx.x;
  if (tid < 128) spm[tid] = spfe[b * 128 + tid] * inv_np;
  __syncthreads();
  if (tid < 32) {
    float a = 0.f, c2 = 0.f;
    for (int k = 0; k < 128; ++k) {
      a  += wg1[tid * 128 + k] * spm[k];
      c2 += ag1[tid * 128 + k] * spm[k];
    }
    t1[tid] = fmaxf(a, 0.f);
    ta[tid] = fmaxf(c2, 0.f);
  }
  __syncthreads();
  for (int o = tid; o < 1152; o += 256) {
    float a = 0.f;
    for (int k = 0; k < 32; ++k) a += wg2[o * 32 + k] * t1[k];
    cw[b * 1152 + o] = a;
  }
  if (tid < 2) {
    float a = 0.f;
    for (int k = 0; k < 32; ++k) a += ag2[tid * 32 + k] * ta[k];
    attn[b * 2 + tid] = 1.f / (1.f + expf(-a));
  }
}

// Depthwise dynamic 3x3 on x1 & x2 + blended output.
__global__ void final_combine(const float* __restrict__ x1, const float* __restrict__ x2,
                              const float* __restrict__ cw, const float* __restrict__ attn,
                              float* __restrict__ out, int B, int C, int H, int W) {
  size_t i = (size_t)blockIdx.x * blockDim.x + threadIdx.x;
  size_t NP = (size_t)H * W;
  size_t tot = (size_t)B * C * NP;
  if (i >= tot) return;
  int w = (int)(i % W);
  int h = (int)((i / W) % H);
  int c = (int)((i / NP) % C);
  int b = (int)(i / (NP * C));
  const float* k9 = cw + ((size_t)b * C + c) * 9;
  const float* p1 = x1 + ((size_t)b * C + c) * NP;
  const float* p2 = x2 + ((size_t)b * C + c) * NP;
  float d1 = 0.f, d2 = 0.f;
#pragma unroll
  for (int kh = 0; kh < 3; ++kh)
#pragma unroll
    for (int kw = 0; kw < 3; ++kw) {
      int hh = h - 1 + kh, ww = w - 1 + kw;
      if (hh < 0 || hh >= H || ww < 0 || ww >= W) continue;
      float k = k9[kh * 3 + kw];
      d1 += k * p1[(size_t)hh * W + ww];
      d2 += k * p2[(size_t)hh * W + ww];
    }
  float a1 = attn[b * 2 + 0], a2 = attn[b * 2 + 1];
  float o1 = p1[(size_t)h * W + w] + 0.5f * a1 * d1;
  float o2 = p2[(size_t)h * W + w] + 0.5f * a2 * d2;
  out[i] = 0.5f * o1 + 0.5f * o2;
}

// ---------------------------------------------------------------------------
extern "C" void kernel_launch(void* const* d_in, const int* in_sizes, int n_in,
                              void* d_out, int out_size, void* d_ws, size_t ws_size,
                              hipStream_t stream) {
  (void)in_sizes; (void)n_in; (void)out_size; (void)ws_size;
  const int B = 8, C = 128, H = 160, W = 160;
  const int NP = H * W;  // 25600

  // d_in order: rgb, t, then params in dict-insertion order.
  const float* rgb  = (const float*)d_in[0];
  const float* t    = (const float*)d_in[1];
  const float* m0_w = (const float*)d_in[2];
  const float* m0_g = (const float*)d_in[3];
  const float* m0_b = (const float*)d_in[4];
  const float* m0_m = (const float*)d_in[5];
  const float* m0_v = (const float*)d_in[6];
  const float* m1_w = (const float*)d_in[7];
  const float* m1_g = (const float*)d_in[8];
  const float* m1_b = (const float*)d_in[9];
  const float* m1_m = (const float*)d_in[10];
  const float* m1_v = (const float*)d_in[11];
  const float* m2_w = (const float*)d_in[12];
  const float* m2_cb = (const float*)d_in[13];
  const float* m2_g = (const float*)d_in[14];
  const float* m2_b = (const float*)d_in[15];
  const float* m2_m = (const float*)d_in[16];
  const float* m2_v = (const float*)d_in[17];
  const float* m3_w = (const float*)d_in[18];
  const float* m3_cb = (const float*)d_in[19];
  const float* m3_g = (const float*)d_in[20];
  const float* m3_b = (const float*)d_in[21];
  const float* m3_m = (const float*)d_in[22];
  const float* m3_v = (const float*)d_in[23];
  const float* m4_w = (const float*)d_in[24];
  const float* m4_cb = (const float*)d_in[25];
  const float* m4_g = (const float*)d_in[26];
  const float* m4_b = (const float*)d_in[27];
  const float* m4_m = (const float*)d_in[28];
  const float* m4_v = (const float*)d_in[29];
  const float* m5_w = (const float*)d_in[30];
  const float* m5_cb = (const float*)d_in[31];
  const float* m5_g = (const float*)d_in[32];
  const float* m5_b = (const float*)d_in[33];
  const float* m5_m = (const float*)d_in[34];
  const float* m5_v = (const float*)d_in[35];
  const float* xy_w = (const float*)d_in[36];
  const float* xy_b = (const float*)d_in[37];
  const float* s1_w = (const float*)d_in[38];
  const float* s1_b = (const float*)d_in[39];
  const float* s2_w = (const float*)d_in[40];
  const float* s2_b = (const float*)d_in[41];
  const float* fus_w = (const float*)d_in[42];
  const float* fus_g = (const float*)d_in[43];
  const float* fus_b = (const float*)d_in[44];
  const float* fus_m = (const float*)d_in[45];
  const float* fus_v = (const float*)d_in[46];
  const float* fe_w = (const float*)d_in[47];
  const float* fe_g = (const float*)d_in[48];
  const float* fe_b = (const float*)d_in[49];
  const float* fe_m = (const float*)d_in[50];
  const float* fe_v = (const float*)d_in[51];
  const float* wg1_w = (const float*)d_in[52];
  const float* wg2_w = (const float*)d_in[53];
  const float* ag1_w = (const float*)d_in[54];
  const float* ag2_w = (const float*)d_in[55];

  // ---- workspace carve (~450 MB) ----
  char* wsp = (char*)d_ws;
  size_t off = 0;
  auto alloc = [&](size_t bytes) -> void* {
    void* p = wsp + off;
    off = (off + bytes + 255) & ~(size_t)255;
    return p;
  };
  _Float16* wpk_m0  = (_Float16*)alloc((size_t)128 * 9 * 256 * 2);
  _Float16* wpk_m1  = (_Float16*)alloc((size_t)128 * 128 * 2);
  _Float16* wpk_m2  = (_Float16*)alloc((size_t)64 * 9 * 128 * 2);
  _Float16* wpk_m3  = (_Float16*)alloc((size_t)64 * 64 * 2);
  _Float16* wpk_m4  = (_Float16*)alloc((size_t)32 * 9 * 64 * 2);
  _Float16* wpk_m5  = (_Float16*)alloc((size_t)32 * 32 * 2);
  _Float16* wpk_fus = (_Float16*)alloc((size_t)128 * 256 * 2);
  _Float16* wpk_fe  = (_Float16*)alloc((size_t)128 * 256 * 2);
  _Float16* in_nhwc = (_Float16*)alloc((size_t)B * NP * 256 * 2);
  _Float16* h0      = (_Float16*)alloc((size_t)B * 6400 * 128 * 2);
  _Float16* h1      = (_Float16*)alloc((size_t)B * 6400 * 128 * 2);
  _Float16* h2      = (_Float16*)alloc((size_t)B * 1600 * 64 * 2);
  _Float16* h3      = (_Float16*)alloc((size_t)B * 1600 * 64 * 2);
  _Float16* h4      = (_Float16*)alloc((size_t)B * 400 * 32 * 2);
  float*    sp5     = (float*)alloc((size_t)B * 32 * 4);
  float*    theta   = (float*)alloc((size_t)B * 4 * 4);
  _Float16* wrapcat = (_Float16*)alloc((size_t)B * NP * 256 * 2);
  float*    x1      = (float*)alloc((size_t)B * C * NP * 4);
  _Float16* fecat   = (_Float16*)alloc((size_t)B * NP * 256 * 2);
  float*    spfe    = (float*)alloc((size_t)B * 128 * 4);
  float*    cwb     = (float*)alloc((size_t)B * 1152 * 4);
  float*    attn    = (float*)alloc((size_t)B * 2 * 4);

  auto g1 = [](long long n) { return (unsigned)((n + 255) / 256); };

  // Weight repacks (f32 -> packed f16).
  repack_w<<<g1(128 * 9 * 256), 256, 0, stream>>>(m0_w, wpk_m0, 128, 256, 9);
  repack_w<<<g1(128 * 128), 256, 0, stream>>>(m1_w, wpk_m1, 128, 128, 1);
  repack_w<<<g1(64 * 9 * 128), 256, 0, stream>>>(m2_w, wpk_m2, 64, 128, 9);
  repack_w<<<g1(64 * 64), 256, 0, stream>>>(m3_w, wpk_m3, 64, 64, 1);
  repack_w<<<g1(32 * 9 * 64), 256, 0, stream>>>(m4_w, wpk_m4, 32, 64, 9);
  repack_w<<<g1(32 * 32), 256, 0, stream>>>(m5_w, wpk_m5, 32, 32, 1);
  repack_w<<<g1(128 * 256), 256, 0, stream>>>(fus_w, wpk_fus, 128, 256, 1);
  repack_w<<<g1(128 * 256), 256, 0, stream>>>(fe_w, wpk_fe, 128, 256, 1);

  pack_input<<<g1((long long)B * C * NP), 256, 0, stream>>>(rgb, t, in_nhwc, wrapcat, fecat, B, C, NP);

  hipMemsetAsync(sp5, 0, (size_t)B * 32 * 4, stream);
  hipMemsetAsync(spfe, 0, (size_t)B * 128 * 4, stream);

  // STN trunk.
  // m0: 3x3 s2, 256->128, 160x160 -> 80x80 (NT=4: 6400/64=100 tiles).
  conv_wmma_direct<4><<<dim3(100, B), 256, 0, stream>>>(in_nhwc, wpk_m0, nullptr,
      m0_g, m0_b, m0_m, m0_v, h0, nullptr, nullptr, 160, 160, 256, 128, 3, 2, 1, 128, 1);
  // m1: 1x1, 128->128 @ 80x80 (async GEMM, 6400/64=100 tiles).
  gemm_wmma_async<<<dim3(100, B), 256, 0, stream>>>(h0, wpk_m1,
      m1_g, m1_b, m1_m, m1_v, h1, nullptr, nullptr, 6400, 128, 128, 128, 1);
  // m2: 3x3 s2, 128->64, 80x80 -> 40x40 (NT=4: 1600/64=25 tiles).
  conv_wmma_direct<4><<<dim3(25, B), 128, 0, stream>>>(h1, wpk_m2, m2_cb,
      m2_g, m2_b, m2_m, m2_v, h2, nullptr, nullptr, 80, 80, 128, 64, 3, 2, 1, 64, 0);
  // m3/m5 carry conv biases -> direct kernel (cb support).
  conv_wmma_direct<1><<<dim3(100, B), 128, 0, stream>>>(h2, wpk_m3, m3_cb,
      m3_g, m3_b, m3_m, m3_v, h3, nullptr, nullptr, 40, 40, 64, 64, 1, 1, 0, 64, 0);
  // m4: 3x3 s2, 64->32, 40x40 -> 20x20 (NP=400, NT=1).
  conv_wmma_direct<1><<<dim3(25, B), 64, 0, stream>>>(h3, wpk_m4, m4_cb,
      m4_g, m4_b, m4_m, m4_v, h4, nullptr, nullptr, 40, 40, 64, 32, 3, 2, 1, 32, 0);
  // m5: 1x1 32->32 @ 20x20, only spatial sums needed.
  conv_wmma_direct<1><<<dim3(25, B), 64, 0, stream>>>(h4, wpk_m5, m5_cb,
      m5_g, m5_b, m5_m, m5_v, nullptr, nullptr, sp5, 20, 20, 32, 32, 1, 1, 0, 32, 0);

  stn_head<<<1, 32, 0, stream>>>(sp5, xy_w, xy_b, s1_w, s1_b, s2_w, s2_b,
                                 theta, B, 32, 1.f / 400.f);

  grid_sample_k<<<g1((long long)B * NP), 256, 0, stream>>>(t, theta, wrapcat, B, C, H, W);

  // fus: x1 (f32 NCHW) + f16 copy into fecat ch 0..127 (stride 256).
  gemm_wmma_async<<<dim3(400, B), 256, 0, stream>>>(wrapcat, wpk_fus,
      fus_g, fus_b, fus_m, fus_v, fecat, x1, nullptr, NP, 256, 128, 256, 1);
  // fe: only the spatial mean is needed -> accumulate sums in epilogue.
  gemm_wmma_async<<<dim3(400, B), 256, 0, stream>>>(fecat, wpk_fe,
      fe_g, fe_b, fe_m, fe_v, nullptr, nullptr, spfe, NP, 256, 128, 1, 0);

  head2<<<B, 256, 0, stream>>>(spfe, wg1_w, wg2_w, ag1_w, ag2_w, cwb, attn,
                               1.f / (float)NP);

  final_combine<<<g1((long long)B * C * NP), 256, 0, stream>>>(x1, t, cwb, attn,
      (float*)d_out, B, C, H, W);
}
